// TypeEncoderLayer_31396210934109
// MI455X (gfx1250) — matrix operand order repaired and checked by reading, not compiled, not run on hardware
//
#include <hip/hip_runtime.h>

// ---------------------------------------------------------------------------
// Problem constants (from the reference)
// ---------------------------------------------------------------------------
#define NS 8
#define NE 48
#define NT 64
#define NG 16
#define TG 128
#define DM 512
#define HH 8
#define DA 32
#define DH 64
#define EPSF 1e-6f

typedef __attribute__((ext_vector_type(16))) __bf16 v16bf;
typedef __attribute__((ext_vector_type(8)))  float  v8f;

__device__ __forceinline__ float bf2f(__bf16 x) { return (float)x; }
__device__ __forceinline__ __bf16 f2bf(float x) { return (__bf16)x; }

__device__ __forceinline__ float wave_sum(float v) {
#pragma unroll
    for (int off = 16; off > 0; off >>= 1) v += __shfl_xor(v, off, 32);
    return v;
}

// Branch-free fast exp: one v_exp_f32.
__device__ __forceinline__ float fexpf(float x) {
    return __builtin_amdgcn_exp2f(x * 1.4426950408889634f);
}

__device__ __forceinline__ float elu1f(float t) {
    return t > 0.f ? t + 1.f : fexpf(t);  // cndmask, no exec branching
}

// tanh-gelu via sigmoid: gelu(x) = x * sigmoid(1.59577*(x + 0.044715 x^3)).
// One v_exp_f32 + one v_rcp_f32, branch-free.
__device__ __forceinline__ float geluf(float x) {
    float u2 = 1.5957691216057308f * (x + 0.044715f * x * x * x);
    return x * __builtin_amdgcn_rcpf(1.f + fexpf(-u2));
}

__device__ __forceinline__ v8f vzero8() {
    v8f z = {0.f, 0.f, 0.f, 0.f, 0.f, 0.f, 0.f, 0.f};
    return z;
}

// C/D f32 16x16 layout: VGPR r holds M=r (lanes 0..15, N=lane) and M=r+8
// (lanes 16..31, N=lane-16).
__device__ __forceinline__ void cpos(int r, int& row, int& col) {
    const int lane = threadIdx.x & 31;
    row = r + ((lane >> 4) << 3);
    col = lane & 15;
}

// ---------------------------------------------------------------------------
// 2-D register-blocked WMMA: MT M-tiles x 2 N-tiles. Per 32-wide K step:
// load 2 B fragments (adjacent 16-col tiles of pre-transposed BT[N][K]) and
// MT A fragments; issue 2*MT v_wmma with full A and B register reuse.
// Fragment k order used here (contiguous, not the ISA table order; A and B match), wave32:
//   A 16-bit 16x32 : lane m(0..15) K=0..15 ; lane m+16 K=16..31
//   B 16-bit 32x16 : lanes 0..15 col n, K=0..15 ; lanes 16..31 col n, K=16..31
// ---------------------------------------------------------------------------
template <int MT>
__device__ __forceinline__ void wmma_mn2(const __bf16* __restrict__ A, int lda,
                                         const __bf16* __restrict__ BT, int ldb,
                                         int ksteps, v8f (*acc)[2]) {
    const int lane = threadIdx.x & 31;
    const int m    = lane & 15;
    const int ka0  = (lane < 16) ? 0 : 16;  // K base for A
    const int nb   = lane & 15;
    const int kb0  = (lane < 16) ? 0 : 16;  // K base for B
    const __bf16* BT1 = BT + 16 * ldb;
    for (int ks = 0; ks < ksteps; ++ks) {
        const int kk = ks * 32;
        v16bf b0, b1;
        const __bf16* pb0 = BT  + nb * ldb + kk + kb0;
        const __bf16* pb1 = BT1 + nb * ldb + kk + kb0;
#pragma unroll
        for (int i = 0; i < 16; ++i) { b0[i] = pb0[i]; b1[i] = pb1[i]; }
#pragma unroll
        for (int mi = 0; mi < MT; ++mi) {
            v16bf a;
            const __bf16* pa0 = A + (mi * 16 + m) * lda + kk + ka0;
#pragma unroll
            for (int i = 0; i < 16; ++i) { a[i] = pa0[i]; }
            acc[mi][0] = __builtin_amdgcn_wmma_f32_16x16x32_bf16(
                false, a, false, b0, (short)0, acc[mi][0], false, false);
            acc[mi][1] = __builtin_amdgcn_wmma_f32_16x16x32_bf16(
                false, a, false, b1, (short)0, acc[mi][1], false, false);
        }
    }
}

// ---------------------------------------------------------------------------
// Weight prep: fp32 [K][N] -> bf16 transposed [N][K]
// ---------------------------------------------------------------------------
__global__ void transpose_bf16_kernel(const float* __restrict__ W,
                                      __bf16* __restrict__ WT, int K, int N) {
    int i = blockIdx.x * blockDim.x + threadIdx.x;
    if (i < K * N) {
        int k = i / N, n = i - k * N;
        WT[(size_t)n * K + k] = f2bf(W[i]);
    }
}

// ---------------------------------------------------------------------------
// Encoder layer: one block (256 thr = 8 wave32) per sequence row; templated
// on S so all tile counts are compile-time.
// LDS layout (bytes), sized for S up to 128:
//   XN  @ 0       : S*512 bf16   -- later reused as ATT, then G32 (32x2048)
//   Q   @ 131072  : S*256 bf16   -- later (with KT) reused as MHA (S*512)
//   KT  @ 196608  : 256*S bf16
//   VCT @ 262144  : 512*32 bf16  -- later KVT [8][64][32], then HN32 (32x512)
//   Z   @ 294912  : 256 f32
//   INV @ 295936  : 128 f32
//   DEN @ 296448  : 128*8 f32
// total 300544 bytes (< 320KB WGP LDS)
// ---------------------------------------------------------------------------
#define ENC_SMEM 300544

template <int S>
__global__ void __launch_bounds__(256) encoder_kernel(
    const float* __restrict__ x,            // [nrows, S, 512]
    const unsigned char* __restrict__ mask, // [nrows, S]
    float* __restrict__ out,                // [nrows, S, 512]
    const float* __restrict__ rms_w,
    const float* __restrict__ ffn_nw,
    const __bf16* __restrict__ wqT,  // [256][512]
    const __bf16* __restrict__ wkT,  // [256][512]
    const __bf16* __restrict__ wvT,  // [512][512]
    const __bf16* __restrict__ woT,  // [512][512]
    const __bf16* __restrict__ w1T,  // [2048][512]
    const __bf16* __restrict__ w2T)  // [512][2048]
{
    constexpr int MTILES = S / 16;
    extern __shared__ char smem[];
    __bf16* XN  = (__bf16*)(smem);
    __bf16* Q   = (__bf16*)(smem + 131072);
    __bf16* KT  = (__bf16*)(smem + 196608);
    __bf16* VCT = (__bf16*)(smem + 262144);
    float*  Z   = (float*)(smem + 294912);
    float*  INV = (float*)(smem + 295936);
    float*  DEN = (float*)(smem + 296448);

    const int row  = blockIdx.x;
    const float* xrow = x + (size_t)row * S * DM;
    float*       orow = out + (size_t)row * S * DM;
    const int tid  = threadIdx.x;
    const int wave = tid >> 5;
    const int lane = tid & 31;

    // ---- Phase 1: xn = rms(x) * rms_w, bf16 to LDS; keep invrms ----
    for (int s = wave; s < S; s += 8) {
        float vals[16];
        float ss = 0.f;
#pragma unroll
        for (int i = 0; i < 16; ++i) {
            float v = xrow[s * DM + lane * 16 + i];
            vals[i] = v;
            ss += v * v;
        }
        ss = wave_sum(ss);
        float inv = rsqrtf(ss / (float)DM + EPSF);
        if (lane == 0) INV[s] = inv;
#pragma unroll
        for (int i = 0; i < 16; ++i) {
            int d = lane * 16 + i;
            XN[s * DM + d] = f2bf(vals[i] * inv * rms_w[d]);
        }
    }
    __syncthreads();

    // ---- Phase 2: Q = elu1(xn @ wq) ; KT[n][s] = elu1(xn @ wk)*mask ----
    // 16 N-tiles = 8 pairs; one pair per wave, MTILES x 2 register block.
    {
        const int ntp = wave;
        v8f acc[MTILES][2];
#pragma unroll
        for (int mi = 0; mi < MTILES; ++mi) { acc[mi][0] = vzero8(); acc[mi][1] = vzero8(); }
        wmma_mn2<MTILES>(XN, DM, wqT + ntp * 32 * 512, 512, 16, acc);
#pragma unroll
        for (int mi = 0; mi < MTILES; ++mi)
#pragma unroll
            for (int ni = 0; ni < 2; ++ni)
#pragma unroll
                for (int r = 0; r < 8; ++r) {
                    int rr, cc; cpos(r, rr, cc);
                    Q[(mi * 16 + rr) * 256 + (ntp * 2 + ni) * 16 + cc] =
                        f2bf(elu1f(acc[mi][ni][r]));
                }
    }
    {
        const int ntp = wave;
        v8f acc[MTILES][2];
#pragma unroll
        for (int mi = 0; mi < MTILES; ++mi) { acc[mi][0] = vzero8(); acc[mi][1] = vzero8(); }
        wmma_mn2<MTILES>(XN, DM, wkT + ntp * 32 * 512, 512, 16, acc);
#pragma unroll
        for (int mi = 0; mi < MTILES; ++mi)
#pragma unroll
            for (int ni = 0; ni < 2; ++ni)
#pragma unroll
                for (int r = 0; r < 8; ++r) {
                    int rr, cc; cpos(r, rr, cc);
                    int s = mi * 16 + rr;
                    float kv = elu1f(acc[mi][ni][r]) * (float)mask[(size_t)row * S + s];
                    KT[((ntp * 2 + ni) * 16 + cc) * S + s] = f2bf(kv);
                }
    }
    __syncthreads();

    // ---- z[d] = sum_s k ----
    {
        int d = tid;  // 256 threads, 256 features
        float zz = 0.f;
        for (int s = 0; s < S; ++s) zz += bf2f(KT[d * S + s]);
        Z[d] = zz;
    }
    __syncthreads();

    // ---- den[s][h] = eps + q . z ----
    for (int e = tid; e < S * HH; e += 256) {
        int s = e >> 3, h = e & 7;
        float dd = EPSF;
        for (int d2 = 0; d2 < DA; ++d2)
            dd += bf2f(Q[s * 256 + h * DA + d2]) * Z[h * DA + d2];
        DEN[e] = dd;
    }

    // ---- Phase 3: kv = K^T V per head (wave==head), regs accumulate ----
    v8f kvacc[2][2][2];  // [n-pair over dh][mt over DA][ni]
#pragma unroll
    for (int i = 0; i < 2; ++i)
#pragma unroll
        for (int j = 0; j < 2; ++j) {
            kvacc[i][j][0] = vzero8();
            kvacc[i][j][1] = vzero8();
        }

    const int h = wave;
    constexpr int NCHUNKS = S / 32;
    for (int cch = 0; cch < NCHUNKS; ++cch) {
        __syncthreads();  // protect VCT reuse
        // V chunk (32 tokens x 512) -> transposed VCT[n][k], ldb=32
        for (int ntp = wave; ntp < 16; ntp += 8) {
            v8f acc[2][2] = {{vzero8(), vzero8()}, {vzero8(), vzero8()}};
            wmma_mn2<2>(XN + (cch * 32) * DM, DM, wvT + ntp * 32 * 512, 512, 16, acc);
#pragma unroll
            for (int mi = 0; mi < 2; ++mi)
#pragma unroll
                for (int ni = 0; ni < 2; ++ni)
#pragma unroll
                    for (int r = 0; r < 8; ++r) {
                        int rr, cc; cpos(r, rr, cc);
                        VCT[((ntp * 2 + ni) * 16 + cc) * 32 + (mi * 16 + rr)] =
                            f2bf(acc[mi][ni][r]);
                    }
        }
        __syncthreads();
        // kv_h += KT_h[:, chunk] @ Vchunk_h
#pragma unroll
        for (int ntp = 0; ntp < 2; ++ntp)
            wmma_mn2<2>(KT + (h * 32) * S + cch * 32, S,
                        VCT + (h * 64 + ntp * 32) * 32, 32, 1, kvacc[ntp]);
    }
    __syncthreads();

    // ---- write kv transposed: KVT[h][e][d] (reuse VCT region) ----
    __bf16* KVT = VCT;  // [8][64][32]
#pragma unroll
    for (int ntp = 0; ntp < 2; ++ntp)
#pragma unroll
        for (int mt2 = 0; mt2 < 2; ++mt2)
#pragma unroll
            for (int ni = 0; ni < 2; ++ni) {
                v8f c = kvacc[ntp][mt2][ni];
#pragma unroll
                for (int r = 0; r < 8; ++r) {
                    int rr, cc; cpos(r, rr, cc);
                    KVT[(h * 64 + ntp * 32 + ni * 16 + cc) * 32 + (mt2 * 16 + rr)] =
                        f2bf(c[r]);
                }
            }
    __syncthreads();

    // ---- Phase 4: att = (Q @ kv) / den -> ATT (reuse XN region) ----
    __bf16* ATT = XN;
    for (int t = wave; t < 16; t += 8) {  // t = (head, n-pair)
        int h2 = t >> 1, np = t & 1;
        v8f acc[MTILES][2];
#pragma unroll
        for (int mi = 0; mi < MTILES; ++mi) { acc[mi][0] = vzero8(); acc[mi][1] = vzero8(); }
        wmma_mn2<MTILES>(Q + h2 * DA, 256, KVT + (h2 * 64 + np * 32) * 32, 32, 1, acc);
#pragma unroll
        for (int mi = 0; mi < MTILES; ++mi)
#pragma unroll
            for (int ni = 0; ni < 2; ++ni)
#pragma unroll
                for (int r = 0; r < 8; ++r) {
                    int rr, cc; cpos(r, rr, cc);
                    int s = mi * 16 + rr;
                    ATT[s * DM + h2 * 64 + (np * 2 + ni) * 16 + cc] =
                        f2bf(acc[mi][ni][r] / DEN[s * 8 + h2]);
                }
    }
    __syncthreads();

    // ---- Phase 5: mha_x = xn + ATT @ wo -> MHA (reuse Q..KT region) ----
    __bf16* MHA = Q;  // S*512 bf16 spans Q+KT regions
    for (int ntp = wave; ntp < 16; ntp += 8) {
        v8f acc[MTILES][2];
#pragma unroll
        for (int mi = 0; mi < MTILES; ++mi) { acc[mi][0] = vzero8(); acc[mi][1] = vzero8(); }
        wmma_mn2<MTILES>(ATT, DM, woT + ntp * 32 * 512, 512, 16, acc);
#pragma unroll
        for (int mi = 0; mi < MTILES; ++mi)
#pragma unroll
            for (int ni = 0; ni < 2; ++ni)
#pragma unroll
                for (int r = 0; r < 8; ++r) {
                    int rr, cc; cpos(r, rr, cc);
                    int s = mi * 16 + rr, d3 = (ntp * 2 + ni) * 16 + cc;
                    float xnv = xrow[s * DM + d3] * INV[s] * rms_w[d3];
                    MHA[s * DM + d3] = f2bf(xnv + acc[mi][ni][r]);
                }
    }
    __syncthreads();

    // ---- Phase 6: FFN in 32-token macro-tiles; out = xn + mha_x + ffn ----
    __bf16* G32  = XN;   // 32*2048 bf16 = 128KB (ATT dead)
    __bf16* HN32 = VCT;  // 32*512 bf16 = 32KB (KVT dead)
    for (int tc = 0; tc < NCHUNKS; ++tc) {
        // h = rms(mha_x) * ffn_nw for 32 tokens
        for (int sl = wave; sl < 32; sl += 8) {
            int s = tc * 32 + sl;
            float vals[16];
            float ss = 0.f;
#pragma unroll
            for (int i = 0; i < 16; ++i) {
                float v = bf2f(MHA[s * DM + lane * 16 + i]);
                vals[i] = v;
                ss += v * v;
            }
            ss = wave_sum(ss);
            float inv = rsqrtf(ss / (float)DM + EPSF);
#pragma unroll
            for (int i = 0; i < 16; ++i) {
                int d = lane * 16 + i;
                HN32[sl * DM + d] = f2bf(vals[i] * inv * ffn_nw[d]);
            }
        }
        __syncthreads();
        // G = gelu(HN @ w1)   (32 x 2048, K=512), 2x2 register block
        for (int ntp = wave; ntp < 64; ntp += 8) {
            v8f acc[2][2] = {{vzero8(), vzero8()}, {vzero8(), vzero8()}};
            wmma_mn2<2>(HN32, DM, w1T + ntp * 32 * 512, 512, 16, acc);
#pragma unroll
            for (int mi = 0; mi < 2; ++mi)
#pragma unroll
                for (int ni = 0; ni < 2; ++ni)
#pragma unroll
                    for (int r = 0; r < 8; ++r) {
                        int rr, cc; cpos(r, rr, cc);
                        G32[(mi * 16 + rr) * 2048 + (ntp * 2 + ni) * 16 + cc] =
                            f2bf(geluf(acc[mi][ni][r]));
                    }
        }
        __syncthreads();
        // out = xn + mha_x + G @ w2   (32 x 512, K=2048)
        for (int ntp = wave; ntp < 16; ntp += 8) {
            v8f acc[2][2] = {{vzero8(), vzero8()}, {vzero8(), vzero8()}};
            wmma_mn2<2>(G32, 2048, w2T + ntp * 32 * 2048, 2048, 64, acc);
#pragma unroll
            for (int mi = 0; mi < 2; ++mi)
#pragma unroll
                for (int ni = 0; ni < 2; ++ni)
#pragma unroll
                    for (int r = 0; r < 8; ++r) {
                        int rr, cc; cpos(r, rr, cc);
                        int s = tc * 32 + mi * 16 + rr, d3 = (ntp * 2 + ni) * 16 + cc;
                        float xnv = xrow[s * DM + d3] * INV[s] * rms_w[d3];
                        orow[s * DM + d3] = xnv + bf2f(MHA[s * DM + d3]) + acc[mi][ni][r];
                    }
        }
        __syncthreads();
    }
}

// ---------------------------------------------------------------------------
// Token-0 update: tu = sf[token0] + noise; sf[token0] = tu
// ---------------------------------------------------------------------------
__global__ void update_kernel(float* __restrict__ sf,
                              const float* __restrict__ noise,
                              float* __restrict__ tu) {
    int r = blockIdx.x;
    for (int d = threadIdx.x; d < DM; d += blockDim.x) {
        float v = sf[(size_t)r * NT * DM + d] + noise[(size_t)r * DM + d];
        tu[(size_t)r * DM + d] = v;
        sf[(size_t)r * NT * DM + d] = v;
    }
}

// ---------------------------------------------------------------------------
// Reference-token residual FFN: gather 32 rows/block, y = base[idx]+tu[ids],
// y + gelu(rms(y)@w1)@w2, scatter back.
// LDS: Y f32 64KB @0 | HN bf16 32KB @65536 | G bf16 128KB @98304 = 224KB.
// ---------------------------------------------------------------------------
#define REF_SMEM (65536 + 32768 + 131072)

__global__ void __launch_bounds__(256) ref_ffn_kernel(
    float* __restrict__ base,
    const int* __restrict__ idx,
    const int* __restrict__ ids,
    const float* __restrict__ tu,
    const float* __restrict__ ref_nw,
    const __bf16* __restrict__ rw1T,   // [2048][512]
    const __bf16* __restrict__ rw2T) { // [512][2048]
    extern __shared__ char smem[];
    float*  Y  = (float*)(smem);
    __bf16* HN = (__bf16*)(smem + 65536);
    __bf16* G  = (__bf16*)(smem + 98304);

    const int b0   = blockIdx.x * 32;
    const int wave = threadIdx.x >> 5;
    const int lane = threadIdx.x & 31;

    for (int sl = wave; sl < 32; sl += 8) {
        int gi = b0 + sl;
        int irow = idx[gi], urow = ids[gi];
        float vals[16];
        float ss = 0.f;
#pragma unroll
        for (int i = 0; i < 16; ++i) {
            int d = lane * 16 + i;
            float v = base[(size_t)irow * DM + d] + tu[(size_t)urow * DM + d];
            vals[i] = v;
            Y[sl * DM + d] = v;
            ss += v * v;
        }
        ss = wave_sum(ss);
        float inv = rsqrtf(ss / (float)DM + EPSF);
#pragma unroll
        for (int i = 0; i < 16; ++i) {
            int d = lane * 16 + i;
            HN[sl * DM + d] = f2bf(vals[i] * inv * ref_nw[d]);
        }
    }
    __syncthreads();

    for (int ntp = wave; ntp < 64; ntp += 8) {
        v8f acc[2][2] = {{vzero8(), vzero8()}, {vzero8(), vzero8()}};
        wmma_mn2<2>(HN, DM, rw1T + ntp * 32 * 512, 512, 16, acc);
#pragma unroll
        for (int mi = 0; mi < 2; ++mi)
#pragma unroll
            for (int ni = 0; ni < 2; ++ni)
#pragma unroll
                for (int r = 0; r < 8; ++r) {
                    int rr, cc; cpos(r, rr, cc);
                    G[(mi * 16 + rr) * 2048 + (ntp * 2 + ni) * 16 + cc] =
                        f2bf(geluf(acc[mi][ni][r]));
                }
    }
    __syncthreads();

    for (int ntp = wave; ntp < 16; ntp += 8) {
        v8f acc[2][2] = {{vzero8(), vzero8()}, {vzero8(), vzero8()}};
        wmma_mn2<2>(G, 2048, rw2T + ntp * 32 * 2048, 2048, 64, acc);
#pragma unroll
        for (int mi = 0; mi < 2; ++mi)
#pragma unroll
            for (int ni = 0; ni < 2; ++ni)
#pragma unroll
                for (int r = 0; r < 8; ++r) {
                    int rr, cc; cpos(r, rr, cc);
                    int grow = mi * 16 + rr;
                    int irow = idx[b0 + grow];
                    base[(size_t)irow * DM + (ntp * 2 + ni) * 16 + cc] =
                        Y[grow * DM + (ntp * 2 + ni) * 16 + cc] + acc[mi][ni][r];
                }
    }
}

// ---------------------------------------------------------------------------
// Launch
// ---------------------------------------------------------------------------
extern "C" void kernel_launch(void* const* d_in, const int* in_sizes, int n_in,
                              void* d_out, int out_size, void* d_ws, size_t ws_size,
                              hipStream_t stream) {
    (void)in_sizes; (void)n_in; (void)out_size; (void)ws_size;

    const float* scope_x   = (const float*)d_in[0];
    const unsigned char* scope_m = (const unsigned char*)d_in[1];
    const int*   s_ref_idx = (const int*)d_in[2];
    const int*   s_ref_ids = (const int*)d_in[3];
    const float* goal_x    = (const float*)d_in[4];
    const unsigned char* goal_m = (const unsigned char*)d_in[5];
    const int*   g_ref_idx = (const int*)d_in[6];
    const int*   g_ref_ids = (const int*)d_in[7];
    const float* noise     = (const float*)d_in[8];
    const float* rms_w     = (const float*)d_in[9];
    const float* wq        = (const float*)d_in[10];
    const float* wk        = (const float*)d_in[11];
    const float* wv        = (const float*)d_in[12];
    const float* wo        = (const float*)d_in[13];
    const float* ffn_nw    = (const float*)d_in[14];
    const float* w1        = (const float*)d_in[15];
    const float* w2        = (const float*)d_in[16];
    const float* ref_nw    = (const float*)d_in[17];
    const float* rw1       = (const float*)d_in[18];
    const float* rw2       = (const float*)d_in[19];

    // workspace layout (bytes)
    char* ws = (char*)d_ws;
    __bf16* wqT  = (__bf16*)(ws + 0);         // 256*512*2  = 262144
    __bf16* wkT  = (__bf16*)(ws + 262144);    // 262144
    __bf16* wvT  = (__bf16*)(ws + 524288);    // 512*512*2  = 524288
    __bf16* woT  = (__bf16*)(ws + 1048576);   // 524288
    __bf16* w1T  = (__bf16*)(ws + 1572864);   // 2048*512*2 = 2097152
    __bf16* w2T  = (__bf16*)(ws + 3670016);   // 512*2048*2 = 2097152
    __bf16* rw1T = (__bf16*)(ws + 5767168);   // 2097152
    __bf16* rw2T = (__bf16*)(ws + 7864320);   // 2097152
    float*  tu   = (float*)(ws + 9961472);    // 384*512*4  = 786432

    auto tgrid = [](int n) { return (n + 255) / 256; };

    transpose_bf16_kernel<<<tgrid(512 * 256),  256, 0, stream>>>(wq,  wqT,  512,  256);
    transpose_bf16_kernel<<<tgrid(512 * 256),  256, 0, stream>>>(wk,  wkT,  512,  256);
    transpose_bf16_kernel<<<tgrid(512 * 512),  256, 0, stream>>>(wv,  wvT,  512,  512);
    transpose_bf16_kernel<<<tgrid(512 * 512),  256, 0, stream>>>(wo,  woT,  512,  512);
    transpose_bf16_kernel<<<tgrid(512 * 2048), 256, 0, stream>>>(w1,  w1T,  512,  2048);
    transpose_bf16_kernel<<<tgrid(2048 * 512), 256, 0, stream>>>(w2,  w2T,  2048, 512);
    transpose_bf16_kernel<<<tgrid(512 * 2048), 256, 0, stream>>>(rw1, rw1T, 512,  2048);
    transpose_bf16_kernel<<<tgrid(2048 * 512), 256, 0, stream>>>(rw2, rw2T, 2048, 512);

    float* sf = (float*)d_out;                  // [384,64,512]
    float* gf = sf + (size_t)NS * NE * NT * DM; // [128,128,512]

    encoder_kernel<NT><<<NS * NE, 256, ENC_SMEM, stream>>>(
        scope_x, scope_m, sf, rms_w, ffn_nw, wqT, wkT, wvT, woT, w1T, w2T);
    encoder_kernel<TG><<<NS * NG, 256, ENC_SMEM, stream>>>(
        goal_x, goal_m, gf, rms_w, ffn_nw, wqT, wkT, wvT, woT, w1T, w2T);

    update_kernel<<<NS * NE, 256, 0, stream>>>(sf, noise, tu);

    ref_ffn_kernel<<<(NS * NE * 8) / 32, 256, REF_SMEM, stream>>>(
        sf, s_ref_idx, s_ref_ids, tu, ref_nw, rw1T, rw2T);
    ref_ffn_kernel<<<(NS * NG * 16) / 32, 256, REF_SMEM, stream>>>(
        gf, g_ref_idx, g_ref_ids, tu, ref_nw, rw1T, rw2T);
}
